// MultiHeadSelfAttention_83056077570571
// MI455X (gfx1250) — compile-verified, matrix operand order checked by reading, not run on hardware
//
#include <hip/hip_runtime.h>

// ---------------- problem constants (from reference) ----------------
#define DMODEL 1024
#define SEQ    2048
#define BATCH  2
#define HEADS  16
#define HD     64                 // head dim
#define MROWS  (BATCH * SEQ)      // 4096 total query rows

typedef __attribute__((ext_vector_type(16))) _Float16 v16h;
typedef __attribute__((ext_vector_type(8)))  _Float16 v8h;
typedef __attribute__((ext_vector_type(4)))  _Float16 v4h;
typedef __attribute__((ext_vector_type(8)))  float    v8f;
typedef __attribute__((ext_vector_type(4)))  float    v4f;

__device__ __forceinline__ v8f wmma16(v16h a, v16h b, v8f c) {
  // D = A(16x32 f16) * B(32x16 f16) + C(16x16 f32)
  return __builtin_amdgcn_wmma_f32_16x16x32_f16(false, a, false, b, (short)0, c,
                                                false, false);
}

__device__ __forceinline__ v8f v8f_zero() {
  v8f z = {0.f, 0.f, 0.f, 0.f, 0.f, 0.f, 0.f, 0.f};
  return z;
}

// ---------------- kernel 1: f32 -> f16 convert (x) ----------------
__global__ void cvt_f16_kernel(const float* __restrict__ in,
                               _Float16* __restrict__ out, int n) {
  int i = (blockIdx.x * blockDim.x + threadIdx.x) * 4;
  if (i + 3 < n) {
    v4f v = *(const v4f*)(in + i);
    v4h h;
    h.x = (_Float16)v.x; h.y = (_Float16)v.y;
    h.z = (_Float16)v.z; h.w = (_Float16)v.w;
    *(v4h*)(out + i) = h;
  }
}

// ---------------- kernel 2: weight transpose + convert: WT[n][k] = W[k][n] --
__global__ void transpose_cvt_kernel(const float* __restrict__ W,
                                     _Float16* __restrict__ WT) {
  __shared__ float tile[32][33];
  const int bx = blockIdx.x * 32;  // n-tile
  const int by = blockIdx.y * 32;  // k-tile
  const int tx = threadIdx.x;      // 0..31
  const int ty = threadIdx.y;      // 0..7
  for (int r = ty; r < 32; r += 8)
    tile[r][tx] = W[(by + r) * DMODEL + bx + tx];
  __syncthreads();
  for (int r = ty; r < 32; r += 8)
    WT[(size_t)(bx + r) * DMODEL + by + tx] = (_Float16)tile[tx][r];
}

// ---------------- kernel 3: fused QKV projection GEMM ----------------
// Each wave: one 32x64 output tile of Y = x @ W + b (W given transposed).
// B fragments are reused across the two 16-row halves -> 8 WMMA / 12 loads.
// mat 0 -> Q (row-major f16), 1 -> K (row-major f16),
// mat 2 -> V (transposed per head: [bh][d][s])
__global__ void qkv_gemm_kernel(const _Float16* __restrict__ xh,
                                const _Float16* __restrict__ WqT,
                                const _Float16* __restrict__ WkT,
                                const _Float16* __restrict__ WvT,
                                const float* __restrict__ bq,
                                const float* __restrict__ bk,
                                const float* __restrict__ bv,
                                _Float16* __restrict__ Qh,
                                _Float16* __restrict__ Kh,
                                _Float16* __restrict__ Vt) {
  const int lane   = threadIdx.x & 31;
  const int wave   = threadIdx.x >> 5;
  const int tileId = blockIdx.x * 4 + wave;

  const int perMat  = (MROWS / 32) * (DMODEL / 64);  // 2048 tiles per matrix
  const int mat     = tileId / perMat;
  const int rem     = tileId % perMat;
  const int rowTile = rem / (DMODEL / 64);
  const int nGroup  = rem % (DMODEL / 64);

  const _Float16* WT   = (mat == 0) ? WqT : (mat == 1) ? WkT : WvT;
  const float*    bias = (mat == 0) ? bq  : (mat == 1) ? bk  : bv;

  const int lhalf = lane & 15;
  const int lhi   = lane >> 4;       // 0: k 0..15 half, 1: k 16..31 half
  const int row0  = rowTile * 32;

  v8f acc[2][4];
#pragma unroll
  for (int u = 0; u < 2; ++u)
#pragma unroll
    for (int t = 0; t < 4; ++t) acc[u][t] = v8f_zero();

  const _Float16* arow0 = xh + (size_t)(row0 + lhalf) * DMODEL + lhi * 16;
  const _Float16* arow1 = arow0 + (size_t)16 * DMODEL;
  const _Float16* bbase = WT + (size_t)(nGroup * 64 + lhalf) * DMODEL + lhi * 16;

  for (int k = 0; k < DMODEL; k += 32) {
    v16h a0 = *(const v16h*)(arow0 + k);
    v16h a1 = *(const v16h*)(arow1 + k);
#pragma unroll
    for (int t = 0; t < 4; ++t) {
      v16h bf = *(const v16h*)(bbase + (size_t)t * 16 * DMODEL + k);
      acc[0][t] = wmma16(a0, bf, acc[0][t]);
      acc[1][t] = wmma16(a1, bf, acc[1][t]);
    }
  }

  if (mat != 2) {
    _Float16* out = (mat == 0) ? Qh : Kh;
#pragma unroll
    for (int u = 0; u < 2; ++u) {
#pragma unroll
      for (int t = 0; t < 4; ++t) {
        const int n  = nGroup * 64 + t * 16 + lhalf;
        const float bv_ = bias[n];
#pragma unroll
        for (int r = 0; r < 8; ++r) {
          const int m = row0 + u * 16 + r + lhi * 8;
          out[(size_t)m * DMODEL + n] = (_Float16)(acc[u][t][r] + bv_);
        }
      }
    }
  } else {
    // V transposed store: Vt[((b*H+h)*HD + d) * SEQ + s]
    const int b = row0 / SEQ;
#pragma unroll
    for (int u = 0; u < 2; ++u) {
      const int s0 = (row0 % SEQ) + u * 16 + lhi * 8;
#pragma unroll
      for (int t = 0; t < 4; ++t) {
        const int n  = nGroup * 64 + t * 16 + lhalf;  // column in D
        const int h  = n / HD;
        const int d  = n % HD;
        const float bv_ = bias[n];
        v8h hv;
#pragma unroll
        for (int r = 0; r < 8; ++r) hv[r] = (_Float16)(acc[u][t][r] + bv_);
        *(v8h*)(Vt + ((size_t)(b * HEADS + h) * HD + d) * SEQ + s0) = hv;
      }
    }
  }
}

// ---------------- kernel 4: flash attention (causal + padding mask) --------
// Block = 4 waves; each wave: 16 query rows x full hd=64, one (b,h).
__global__ void attn_kernel(const _Float16* __restrict__ Qh,
                            const _Float16* __restrict__ Kh,
                            const _Float16* __restrict__ Vt,
                            const int* __restrict__ amask,
                            _Float16* __restrict__ ctx) {
  __shared__ __align__(128) _Float16 plds[4 * 16 * 32];  // per-wave P staging

  const int lane  = threadIdx.x & 31;
  const int wave  = threadIdx.x >> 5;
  const int lhalf = lane & 15;
  const int lhi   = lane >> 4;

  const int qblk  = blockIdx.x % (SEQ / 64);
  const int bh    = blockIdx.x / (SEQ / 64);
  const int b     = bh / HEADS;
  const int h     = bh % HEADS;
  const int qbase = qblk * 64 + wave * 16;

  _Float16* myP = plds + wave * (16 * 32);

  // resident Q fragments (hd=64 => two 32-wide K steps)
  const _Float16* qrow =
      Qh + (size_t)(b * SEQ + qbase + lhalf) * DMODEL + h * HD + lhi * 16;
  v16h qf0 = *(const v16h*)(qrow);
  v16h qf1 = *(const v16h*)(qrow + 32);

  v8f o[4];
  v8f mrow, lrow;
#pragma unroll
  for (int t = 0; t < 4; ++t) o[t] = v8f_zero();
#pragma unroll
  for (int r = 0; r < 8; ++r) { mrow[r] = -1e30f; lrow[r] = 0.f; }

  const float scale = 0.125f;  // 1/sqrt(64)
  const int nkb = (qbase + 16 + 31) / 32;  // causal: keys <= qbase+15

  const _Float16* vbase0 = Vt + (size_t)(b * HEADS + h) * HD * SEQ;

  for (int kb = 0; kb < nkb; ++kb) {
    const int k0 = kb * 32;

    // ---- scores S = Q Kt : two 16x16 tiles ----
    v8f st[2];
#pragma unroll
    for (int t = 0; t < 2; ++t) {
      const int key = k0 + t * 16 + lhalf;
      const _Float16* krow =
          Kh + (size_t)(b * SEQ + key) * DMODEL + h * HD + lhi * 16;
      v16h b0 = *(const v16h*)(krow);
      v16h b1 = *(const v16h*)(krow + 32);
      v8f c = v8f_zero();
      c = wmma16(qf0, b0, c);
      c = wmma16(qf1, b1, c);
      st[t] = c;
    }

    // ---- scale + padding mask (+ causal on diagonal blocks) ----
#pragma unroll
    for (int t = 0; t < 2; ++t) {
      const int key = k0 + t * 16 + lhalf;
      const float pad = (amask[b * SEQ + key] > 0) ? 0.f : -1e30f;
#pragma unroll
      for (int r = 0; r < 8; ++r) st[t][r] = st[t][r] * scale + pad;
    }
    if (k0 + 31 > qbase) {  // wave-uniform: some kg > some qg
#pragma unroll
      for (int t = 0; t < 2; ++t) {
        const int key = k0 + t * 16 + lhalf;
#pragma unroll
        for (int r = 0; r < 8; ++r) {
          const int qg = qbase + r + lhi * 8;
          if (key > qg) st[t][r] = -1e30f;
        }
      }
    }

    // ---- online softmax: row max across 32 keys ----
    v8f rmax;
#pragma unroll
    for (int r = 0; r < 8; ++r) rmax[r] = fmaxf(st[0][r], st[1][r]);
#pragma unroll
    for (int off = 1; off < 16; off <<= 1) {
#pragma unroll
      for (int r = 0; r < 8; ++r)
        rmax[r] = fmaxf(rmax[r], __shfl_xor(rmax[r], off, 32));
    }

    v8f mnew, alpha;
#pragma unroll
    for (int r = 0; r < 8; ++r) {
      mnew[r]  = fmaxf(mrow[r], rmax[r]);
      alpha[r] = __expf(mrow[r] - mnew[r]);
      mrow[r]  = mnew[r];
    }

    v8f rsum = v8f_zero();
#pragma unroll
    for (int t = 0; t < 2; ++t) {
#pragma unroll
      for (int r = 0; r < 8; ++r) {
        const float p = __expf(st[t][r] - mnew[r]);
        rsum[r] += p;
        myP[(r + lhi * 8) * 32 + t * 16 + lhalf] = (_Float16)p;  // C -> LDS
      }
    }
#pragma unroll
    for (int off = 1; off < 16; off <<= 1) {
#pragma unroll
      for (int r = 0; r < 8; ++r) rsum[r] += __shfl_xor(rsum[r], off, 32);
    }
#pragma unroll
    for (int r = 0; r < 8; ++r) lrow[r] = lrow[r] * alpha[r] + rsum[r];
#pragma unroll
    for (int t = 0; t < 4; ++t)
#pragma unroll
      for (int r = 0; r < 8; ++r) o[t][r] *= alpha[r];

    // ---- P (A-layout via LDS) x V : 4 output tiles over hd ----
    asm volatile("s_wait_dscnt 0" ::: "memory");
    v16h pf = *(const v16h*)(myP + lhalf * 32 + lhi * 16);
    const _Float16* vkp = vbase0 + k0 + lhi * 16;
#pragma unroll
    for (int t = 0; t < 4; ++t) {
      const int d = t * 16 + lhalf;
      v16h vf = *(const v16h*)(vkp + (size_t)d * SEQ);
      o[t] = wmma16(pf, vf, o[t]);
    }
  }

  // ---- normalize + store ctx (row-major f16) ----
  v8f inv;
#pragma unroll
  for (int r = 0; r < 8; ++r) inv[r] = 1.0f / lrow[r];
  _Float16* crow = ctx + (size_t)(b * SEQ + qbase) * DMODEL + h * HD;
#pragma unroll
  for (int t = 0; t < 4; ++t) {
    const int d = t * 16 + lhalf;
#pragma unroll
    for (int r = 0; r < 8; ++r) {
      const int m = r + lhi * 8;
      crow[(size_t)m * DMODEL + d] = (_Float16)(o[t][r] * inv[r]);
    }
  }
}

// ---------------- kernel 5: output projection (f32 out) ----------------
// Each wave: 32x64 output tile with B-fragment reuse (8 WMMA / 12 loads).
__global__ void out_gemm_kernel(const _Float16* __restrict__ ch,
                                const _Float16* __restrict__ WoT,
                                const float* __restrict__ bo,
                                float* __restrict__ out) {
  const int lane   = threadIdx.x & 31;
  const int wave   = threadIdx.x >> 5;
  const int tileId = blockIdx.x * 4 + wave;
  const int rowTile = tileId / (DMODEL / 64);
  const int nGroup  = tileId % (DMODEL / 64);
  const int lhalf = lane & 15;
  const int lhi   = lane >> 4;
  const int row0  = rowTile * 32;

  v8f acc[2][4];
#pragma unroll
  for (int u = 0; u < 2; ++u)
#pragma unroll
    for (int t = 0; t < 4; ++t) acc[u][t] = v8f_zero();

  const _Float16* arow0 = ch + (size_t)(row0 + lhalf) * DMODEL + lhi * 16;
  const _Float16* arow1 = arow0 + (size_t)16 * DMODEL;
  const _Float16* bbase = WoT + (size_t)(nGroup * 64 + lhalf) * DMODEL + lhi * 16;

  for (int k = 0; k < DMODEL; k += 32) {
    v16h a0 = *(const v16h*)(arow0 + k);
    v16h a1 = *(const v16h*)(arow1 + k);
#pragma unroll
    for (int t = 0; t < 4; ++t) {
      v16h bf = *(const v16h*)(bbase + (size_t)t * 16 * DMODEL + k);
      acc[0][t] = wmma16(a0, bf, acc[0][t]);
      acc[1][t] = wmma16(a1, bf, acc[1][t]);
    }
  }

#pragma unroll
  for (int u = 0; u < 2; ++u) {
#pragma unroll
    for (int t = 0; t < 4; ++t) {
      const int n = nGroup * 64 + t * 16 + lhalf;
      const float bb = bo[n];
#pragma unroll
      for (int r = 0; r < 8; ++r)
        out[(size_t)(row0 + u * 16 + r + lhi * 8) * DMODEL + n] =
            acc[u][t][r] + bb;
    }
  }
}

// ---------------- launcher ----------------
extern "C" void kernel_launch(void* const* d_in, const int* in_sizes, int n_in,
                              void* d_out, int out_size, void* d_ws,
                              size_t ws_size, hipStream_t stream) {
  (void)in_sizes; (void)n_in; (void)out_size; (void)ws_size;
  const float* x     = (const float*)d_in[0];
  const int*   amask = (const int*)d_in[1];
  const float* Wq    = (const float*)d_in[2];
  const float* bq    = (const float*)d_in[3];
  const float* Wk    = (const float*)d_in[4];
  const float* bk    = (const float*)d_in[5];
  const float* Wv    = (const float*)d_in[6];
  const float* bv    = (const float*)d_in[7];
  const float* Wo    = (const float*)d_in[8];
  const float* bo    = (const float*)d_in[9];
  float* out = (float*)d_out;

  // carve workspace (all 256B aligned)
  char* ws = (char*)d_ws;
  size_t off = 0;
  auto carve = [&](size_t bytes) {
    void* p = ws + off;
    off += (bytes + 255) & ~(size_t)255;
    return p;
  };
  const size_t actBytes = (size_t)MROWS * DMODEL * sizeof(_Float16);  // 8 MB
  const size_t wBytes   = (size_t)DMODEL * DMODEL * sizeof(_Float16); // 2 MB
  _Float16* xh  = (_Float16*)carve(actBytes);
  _Float16* WqT = (_Float16*)carve(wBytes);
  _Float16* WkT = (_Float16*)carve(wBytes);
  _Float16* WvT = (_Float16*)carve(wBytes);
  _Float16* WoT = (_Float16*)carve(wBytes);
  _Float16* Qh  = (_Float16*)carve(actBytes);
  _Float16* Kh  = (_Float16*)carve(actBytes);
  _Float16* Vt  = (_Float16*)carve(actBytes);
  _Float16* Ch  = (_Float16*)carve(actBytes);

  // 1) convert x to f16
  {
    const int n = MROWS * DMODEL;
    cvt_f16_kernel<<<n / 4 / 256, 256, 0, stream>>>(x, xh, n);
  }
  // 2) transpose+convert weights
  {
    dim3 g(DMODEL / 32, DMODEL / 32), t(32, 8);
    transpose_cvt_kernel<<<g, t, 0, stream>>>(Wq, WqT);
    transpose_cvt_kernel<<<g, t, 0, stream>>>(Wk, WkT);
    transpose_cvt_kernel<<<g, t, 0, stream>>>(Wv, WvT);
    transpose_cvt_kernel<<<g, t, 0, stream>>>(Wo, WoT);
  }
  // 3) fused QKV projection: 3 * 2048 tiles (32x64 each), 4 waves/block
  qkv_gemm_kernel<<<(3 * (MROWS / 32) * (DMODEL / 64)) / 4, 128, 0, stream>>>(
      xh, WqT, WkT, WvT, bq, bk, bv, Qh, Kh, Vt);
  // 4) flash attention: B*H*(S/64) blocks of 4 waves
  attn_kernel<<<BATCH * HEADS * (SEQ / 64), 128, 0, stream>>>(Qh, Kh, Vt,
                                                              amask, Ch);
  // 5) output projection: 2048 tiles (32x64 each), 4 waves/block
  out_gemm_kernel<<<((MROWS / 32) * (DMODEL / 64)) / 4, 128, 0, stream>>>(
      Ch, WoT, bo, out);
}
